// RecurrentModelWithCRF_36275293782555
// MI455X (gfx1250) — matrix operand order repaired and checked by reading, not compiled, not run on hardware
//
#include <hip/hip_runtime.h>
#include <hip/hip_bf16.h>
#include <math.h>

typedef __attribute__((ext_vector_type(16))) __bf16 v16bf;
typedef __attribute__((ext_vector_type(8)))  __bf16 v8bf;
typedef __attribute__((ext_vector_type(8)))  float  v8f;

#define T_LEN 4096
#define BATCH 16
#define DIN   256
#define HID   256
#define G4    1024   // 4*HID
#define NTAG  3

struct U16B2 { uint4 a; uint4 b; };   // 32 bytes == v16bf

__device__ __forceinline__ v16bf pack16(uint4 lo, uint4 hi) {
  U16B2 u; u.a = lo; u.b = hi;
  return __builtin_bit_cast(v16bf, u);
}

// ---------------------------------------------------------------------------
// Kernel A: xproj[t][col][b] = (x[b][t][:] @ Wih.T + bias), stored in WMMA
// C-matrix order [T, G4, B] bf16 so the recurrence reads one b128 per tile.
// One wave per 16x16 output tile; K=256 -> 8 WMMA bf16 ops.
// ---------------------------------------------------------------------------
__global__ __launch_bounds__(32) void xproj_kernel(
    const float* __restrict__ x,            // [B,T,D]
    const float* __restrict__ WihF, const float* __restrict__ bF,
    const float* __restrict__ WihB, const float* __restrict__ bB,
    __bf16* __restrict__ xpF,               // [T,G4,B] bf16
    __bf16* __restrict__ xpB)
{
  const int t   = blockIdx.x;
  const int n0  = blockIdx.y * 16;
  const int dir = blockIdx.z;
  const float* W    = dir ? WihB : WihF;
  const float* bias = dir ? bB   : bF;
  __bf16* out = dir ? xpB : xpF;

  const int lane = threadIdx.x;
  const int half = lane >> 4;
  const int l15  = lane & 15;

  v8f acc;
  const float bv = bias[n0 + l15];
  #pragma unroll
  for (int r = 0; r < 8; ++r) acc[r] = bv;

  #pragma unroll
  for (int kc = 0; kc < DIN / 32; ++kc) {
    const int ks = kc * 32;
    // A tile: row M = batch = l15; halves split K 0-15/16-31 (f32 -> bf16 cvt)
    v16bf a;
    {
      const int aoff = half ? 16 : 0;
      const float* xr = x + ((size_t)l15 * T_LEN + t) * DIN + ks;
      #pragma unroll
      for (int e = 0; e < 8; ++e) {
        a[e]     = (__bf16)xr[aoff + e];
        a[8 + e] = (__bf16)xr[8 + aoff + e];
      }
    }
    // B tile: lane = column (W row, contiguous in K); halves split K 0-15/16-31
    v16bf bm;
    {
      const float* wr = W + (size_t)(n0 + l15) * DIN + ks + half * 16;
      #pragma unroll
      for (int e = 0; e < 16; ++e) bm[e] = (__bf16)wr[e];
    }
    acc = __builtin_amdgcn_wmma_f32_16x16x32_bf16(false, a, false, bm,
                                                  (short)0, acc, false, false);
  }

  // Store in C-layout order: 8 contiguous bf16 (rows 8*half..8*half+7) per lane.
  const int col = n0 + l15;
  v8bf ov;
  #pragma unroll
  for (int r = 0; r < 8; ++r) ov[r] = (__bf16)acc[r];
  *(v8bf*)(out + ((size_t)t * G4 + col) * BATCH + 8 * half) = ov;
}

// ---------------------------------------------------------------------------
// Kernel B: sequential LSTM recurrence, one workgroup per direction.
// 32 waves; wave w owns gate columns [32w, 32w+32). Whh held in VGPRs as
// pre-packed WMMA B operands (2 tiles x 8 K-chunks). h in LDS (bf16),
// gates staged through LDS in [col][b] order, c kept in thread registers.
// ---------------------------------------------------------------------------
__global__ __launch_bounds__(1024) void lstm_kernel(
    const float* __restrict__ WhhF, const float* __restrict__ WhhB,
    const __bf16* __restrict__ xpF, const __bf16* __restrict__ xpB,
    const int* __restrict__ lengths,
    __bf16* __restrict__ hF, __bf16* __restrict__ hB)
{
  const int dir = blockIdx.x;
  const float* Whh     = dir ? WhhB : WhhF;
  const __bf16* xp     = dir ? xpB  : xpF;
  __bf16* hout         = dir ? hB   : hF;

  __shared__ __bf16 h_lds[BATCH * HID];   // [b][k], 8 KB
  __shared__ __bf16 g_lds[G4 * BATCH];    // [col][b], 32 KB

  const int tid  = threadIdx.x;
  const int lane = tid & 31;
  const int wave = tid >> 5;
  const int half = lane >> 4;
  const int l15  = lane & 15;

  // Preload this wave's Whh slice as packed B operands (stays in VGPRs).
  v16bf wB[2][8];
  #pragma unroll
  for (int tile = 0; tile < 2; ++tile) {
    const int col = wave * 32 + tile * 16 + l15;
    #pragma unroll
    for (int kc = 0; kc < 8; ++kc) {
      const float* wr = Whh + (size_t)col * HID + kc * 32 + half * 16;
      v16bf v;
      #pragma unroll
      for (int e = 0; e < 16; ++e) v[e] = (__bf16)wr[e];
      wB[tile][kc] = v;
    }
  }

  // Elementwise mapping: fixed (b, 4 columns) per thread across all steps.
  const int eb = tid >> 6;
  const int ej = (tid & 63) * 4;
  const int len_eb = lengths[eb];
  float c_st[4] = {0.f, 0.f, 0.f, 0.f};
  float h_st[4] = {0.f, 0.f, 0.f, 0.f};

  for (int i = tid; i < BATCH * HID; i += 1024) h_lds[i] = (__bf16)0.f;
  __syncthreads();

  for (int t = 0; t < T_LEN; ++t) {
    const int time = dir ? (T_LEN - 1 - t) : t;

    // Prefetch next step's xproj slice into the cache hierarchy.
    if (t + 1 < T_LEN && lane == 0) {
      const int nt = dir ? (T_LEN - 2 - t) : (t + 1);
      __builtin_prefetch(xp + ((size_t)nt * G4 + wave * 32) * BATCH, 0, 1);
    }

    // ---- matmul phase: gates = xproj[time] + h @ Whh.T ----
    // A fragments: two contiguous 16B runs per chunk from h_lds (ds_load_b128).
    v16bf a[8];
    #pragma unroll
    for (int kc = 0; kc < 8; ++kc) {
      const int aoff = half ? 16 : 0;
      const __bf16* hr = h_lds + l15 * HID + kc * 32 + aoff;
      const uint4 lo = *(const uint4*)(hr);
      const uint4 hi = *(const uint4*)(hr + 8);
      a[kc] = pack16(lo, hi);
    }
    #pragma unroll
    for (int tile = 0; tile < 2; ++tile) {
      const int col = wave * 32 + tile * 16 + l15;
      // one global_load_b128: 8 contiguous bf16 in C-layout order
      const v8bf xv = *(const v8bf*)(xp + ((size_t)time * G4 + col) * BATCH + 8 * half);
      v8f acc;
      #pragma unroll
      for (int r = 0; r < 8; ++r) acc[r] = (float)xv[r];
      #pragma unroll
      for (int kc = 0; kc < 8; ++kc)
        acc = __builtin_amdgcn_wmma_f32_16x16x32_bf16(false, a[kc], false, wB[tile][kc],
                                                      (short)0, acc, false, false);
      v8bf gv;
      #pragma unroll
      for (int r = 0; r < 8; ++r) gv[r] = (__bf16)acc[r];
      *(v8bf*)(g_lds + col * BATCH + 8 * half) = gv;   // one ds_store_b128
    }
    __syncthreads();

    // ---- elementwise LSTM cell (packed-sequence masking) ----
    const bool m = time < len_eb;
    #pragma unroll
    for (int jj = 0; jj < 4; ++jj) {
      const int j = ej + jj;
      const float gi = (float)g_lds[(j)       * BATCH + eb];
      const float gf = (float)g_lds[(256 + j) * BATCH + eb];
      const float gg = (float)g_lds[(512 + j) * BATCH + eb];
      const float go = (float)g_lds[(768 + j) * BATCH + eb];
      const float si = 1.f / (1.f + __expf(-gi));
      const float sf = 1.f / (1.f + __expf(-gf));
      const float so = 1.f / (1.f + __expf(-go));
      const float cn = sf * c_st[jj] + si * tanhf(gg);
      const float hn = so * tanhf(cn);
      const float ck = m ? cn : c_st[jj];
      const float hk = m ? hn : h_st[jj];
      c_st[jj] = ck;
      h_st[jj] = hk;
      h_lds[eb * HID + j] = (__bf16)hk;
      hout[((size_t)time * BATCH + eb) * HID + j] = (__bf16)(m ? hn : 0.f);
    }
    __syncthreads();
  }
}

// ---------------------------------------------------------------------------
// Kernel C: emissions em = [h_f ; h_b] @ W_tag.T + b_tag   (N=3, scalar dots)
// ---------------------------------------------------------------------------
__global__ __launch_bounds__(256) void em_kernel(
    const __bf16* __restrict__ hF, const __bf16* __restrict__ hB,
    const float* __restrict__ Wtag, const float* __restrict__ btag,
    float* __restrict__ em)
{
  const int idx = blockIdx.x * 256 + threadIdx.x;    // t*B + b
  if (idx >= T_LEN * BATCH) return;
  float a0 = btag[0], a1 = btag[1], a2 = btag[2];
  const __bf16* hf = hF + (size_t)idx * HID;
  const __bf16* hb = hB + (size_t)idx * HID;
  for (int j = 0; j < HID; ++j) {
    const float v = (float)hf[j];
    a0 += v * Wtag[0 * 512 + j];
    a1 += v * Wtag[1 * 512 + j];
    a2 += v * Wtag[2 * 512 + j];
  }
  for (int j = 0; j < HID; ++j) {
    const float v = (float)hb[j];
    a0 += v * Wtag[0 * 512 + 256 + j];
    a1 += v * Wtag[1 * 512 + 256 + j];
    a2 += v * Wtag[2 * 512 + 256 + j];
  }
  em[(size_t)idx * 3 + 0] = a0;
  em[(size_t)idx * 3 + 1] = a1;
  em[(size_t)idx * 3 + 2] = a2;
}

// ---------------------------------------------------------------------------
// Kernel D: CRF Viterbi decode + backtrack. One lane per batch element.
// ---------------------------------------------------------------------------
__global__ __launch_bounds__(32) void crf_kernel(
    const float* __restrict__ em, const int* __restrict__ lengths,
    const float* __restrict__ start_t, const float* __restrict__ end_t,
    const float* __restrict__ trans,
    unsigned char* __restrict__ bp,     // [T-1][B][3]
    float* __restrict__ out)            // paths [B][T] as float, then scores [B]
{
  const int b = threadIdx.x;
  if (b >= BATCH) return;
  const int len = lengths[b];
  float tr[3][3];
  for (int i = 0; i < 3; ++i)
    for (int j = 0; j < 3; ++j) tr[i][j] = trans[i * 3 + j];

  float s0 = start_t[0] + em[(size_t)b * 3 + 0];
  float s1 = start_t[1] + em[(size_t)b * 3 + 1];
  float s2 = start_t[2] + em[(size_t)b * 3 + 2];

  for (int t = 1; t < T_LEN; ++t) {
    const float* e = em + ((size_t)t * BATCH + b) * 3;
    const float sc[3] = {s0, s1, s2};
    float ns[3]; int nb[3];
    #pragma unroll
    for (int cn = 0; cn < 3; ++cn) {
      float best = sc[0] + tr[0][cn]; int bi = 0;
      #pragma unroll
      for (int cp = 1; cp < 3; ++cp) {
        const float v = sc[cp] + tr[cp][cn];
        if (v > best) { best = v; bi = cp; }
      }
      ns[cn] = best + e[cn]; nb[cn] = bi;
    }
    unsigned char* bpt = bp + ((size_t)(t - 1) * BATCH + b) * 3;
    bpt[0] = (unsigned char)nb[0];
    bpt[1] = (unsigned char)nb[1];
    bpt[2] = (unsigned char)nb[2];
    const bool m = t < len;
    s0 = m ? ns[0] : s0; s1 = m ? ns[1] : s1; s2 = m ? ns[2] : s2;
  }

  const float f0 = s0 + end_t[0], f1 = s1 + end_t[1], f2 = s2 + end_t[2];
  float best = f0; int btag = 0;
  if (f1 > best) { best = f1; btag = 1; }
  if (f2 > best) { best = f2; btag = 2; }
  out[(size_t)BATCH * T_LEN + b] = best;

  int cur = btag;
  out[(size_t)b * T_LEN + (T_LEN - 1)] = ((T_LEN - 1) < len) ? (float)cur : 0.f;
  for (int t = T_LEN - 1; t >= 1; --t) {
    if (t < len) cur = bp[((size_t)(t - 1) * BATCH + b) * 3 + cur];
    out[(size_t)b * T_LEN + (t - 1)] = ((t - 1) < len) ? (float)cur : 0.f;
  }
}

// ---------------------------------------------------------------------------
extern "C" void kernel_launch(void* const* d_in, const int* in_sizes, int n_in,
                              void* d_out, int out_size, void* d_ws, size_t ws_size,
                              hipStream_t stream) {
  (void)in_sizes; (void)n_in; (void)out_size; (void)ws_size;
  const float* sentences = (const float*)d_in[0];
  const int*   lengths   = (const int*)d_in[1];
  const float* WihF = (const float*)d_in[2];
  const float* WhhF = (const float*)d_in[3];
  const float* bF   = (const float*)d_in[4];
  const float* WihB = (const float*)d_in[5];
  const float* WhhB = (const float*)d_in[6];
  const float* bB   = (const float*)d_in[7];
  const float* Wtag = (const float*)d_in[8];
  const float* btag = (const float*)d_in[9];
  const float* start_t = (const float*)d_in[10];
  const float* end_t   = (const float*)d_in[11];
  const float* trans   = (const float*)d_in[12];

  char* ws = (char*)d_ws;
  size_t off = 0;
  __bf16* xpF = (__bf16*)(ws + off); off += (size_t)T_LEN * G4 * BATCH * 2;
  __bf16* xpB = (__bf16*)(ws + off); off += (size_t)T_LEN * G4 * BATCH * 2;
  __bf16* hF  = (__bf16*)(ws + off); off += (size_t)T_LEN * BATCH * HID * 2;
  __bf16* hB  = (__bf16*)(ws + off); off += (size_t)T_LEN * BATCH * HID * 2;
  float* em           = (float*)(ws + off);         off += (size_t)T_LEN * BATCH * NTAG * 4;
  unsigned char* bp   = (unsigned char*)(ws + off); off += (size_t)(T_LEN - 1) * BATCH * NTAG;

  xproj_kernel<<<dim3(T_LEN, G4 / 16, 2), 32, 0, stream>>>(
      sentences, WihF, bF, WihB, bB, xpF, xpB);
  lstm_kernel<<<dim3(2), 1024, 0, stream>>>(
      WhhF, WhhB, xpF, xpB, lengths, hF, hB);
  em_kernel<<<dim3((T_LEN * BATCH + 255) / 256), 256, 0, stream>>>(
      hF, hB, Wtag, btag, em);
  crf_kernel<<<dim3(1), 32, 0, stream>>>(
      em, lengths, start_t, end_t, trans, bp, (float*)d_out);
}
